// HigherOrderAttentionLayer_12335146074694
// MI455X (gfx1250) — matrix operand order repaired and checked by reading, not compiled, not run on hardware
//
#include <hip/hip_runtime.h>
#include <hip/hip_bf16.h>

typedef __attribute__((ext_vector_type(16))) _Float16 v16h;
typedef __attribute__((ext_vector_type(8)))  float    v8f;
typedef __attribute__((ext_vector_type(4)))  int      v4i;
typedef __attribute__((ext_vector_type(4)))  float    v4f;

#define NSRC 8192
#define NTGT 4096
#define DIN  128
#define DOUT 64
#define LRA  0.1f
#define NEGV (-9.0e15f)

// Branchless online softmax accumulate: m starts at NEGV, all math finite.
__device__ __forceinline__ void online_ms(float& m, float& s, float e) {
  float nm = fmaxf(m, e);
  s = s * __expf(m - nm) + __expf(e - nm);
  m = nm;
}

// Branchless (m,s) pair merge (valid because m values never go below NEGV).
__device__ __forceinline__ void combine_ms(float& m1, float& s1, float m2, float s2) {
  float nm = fmaxf(m1, m2);
  s1 = s1 * __expf(m1 - nm) + s2 * __expf(m2 - nm);
  m1 = nm;
}

// Branch-free ELU: max(x,0) + exp(min(x,0)) - 1.
__device__ __forceinline__ float elu_bf(float x) {
  return fmaxf(x, 0.f) + (__expf(fminf(x, 0.f)) - 1.f);
}

// ---------------------------------------------------------------------------
// Kernel 0: projections Whs=hs@Ws, Wht=ht@Wt and scalars Whst=Whs@a_s,
// Whtt=Wht@a_t. One 64-thread block per row (first NSRC blocks = hs rows).
// ---------------------------------------------------------------------------
__global__ void gat_proj_kernel(const float* __restrict__ hs, const float* __restrict__ ht,
                                const float* __restrict__ Ws, const float* __restrict__ Wt,
                                const float* __restrict__ a1,
                                float* __restrict__ Whs, float* __restrict__ Wht,
                                float* __restrict__ Whst, float* __restrict__ Whtt) {
  __shared__ float row[DIN];
  __shared__ float red[DOUT];
  const int b = blockIdx.x, c = threadIdx.x;
  const float* in; const float* W; const float* av; float* outW; float* outd;
  if (b < NSRC) {
    int r = b;
    in = hs + (size_t)r * DIN; W = Ws; av = a1 + DOUT;           // a_s = a1[64:]
    outW = Whs + (size_t)r * DOUT; outd = Whst + r;
  } else {
    int r = b - NSRC;
    in = ht + (size_t)r * DIN; W = Wt; av = a1;                  // a_t = a1[:64]
    outW = Wht + (size_t)r * DOUT; outd = Whtt + r;
  }
  row[c] = in[c];
  row[c + 64] = in[c + 64];
  __syncthreads();
  float acc = 0.f;
#pragma unroll 8
  for (int k = 0; k < DIN; ++k) acc += row[k] * W[k * DOUT + c];
  outW[c] = acc;
  red[c] = acc * av[c];
  __syncthreads();
  for (int off = 32; off > 0; off >>= 1) {
    if (c < off) red[c] += red[c + off];
    __syncthreads();
  }
  if (c == 0) *outd = red[0];
}

// ---------------------------------------------------------------------------
// Kernel 1: pack Wh (rows x 64, f32) into f16 WMMA B-fragment order.
// Fragment (kt,nt): lane l holds column N=l&15, slot s holds K=(l>>4)*8+(s&7)+(s>>3)*16.
// Memory index = ((kt*4+nt)*32 + l)*16 + s  -> v16h per lane is contiguous.
// ---------------------------------------------------------------------------
__global__ void gat_pack_kernel(const float* __restrict__ src, _Float16* __restrict__ dst,
                                int rows) {
  int tid = blockIdx.x * blockDim.x + threadIdx.x;
  if (tid >= rows * DOUT) return;
  int s   = tid & 15;
  int l   = (tid >> 4) & 31;
  int ntk = tid >> 9;
  int nt  = ntk & 3;
  int kt  = ntk >> 2;
  int j   = kt * 32 + (l >> 4) * 8 + (s & 7) + (s >> 3) * 16;
  int col = nt * 16 + (l & 15);
  dst[tid] = (_Float16)src[(size_t)j * DOUT + col];
}

// ---------------------------------------------------------------------------
// Kernel 2: online-softmax stats per row of att_st (one block per target i,
// coalesced scan of A row). Masked entries use the reference's NEG logit so
// the loop body is fully branchless (exp is unconditional).
// ---------------------------------------------------------------------------
__global__ void gat_stats_st_kernel(const int* __restrict__ A, const float* __restrict__ Whst,
                                    const float* __restrict__ Whtt,
                                    float* __restrict__ m_st, float* __restrict__ inv_st) {
  __shared__ float lws[NSRC];          // 32 KB of the 320 KB WGP LDS
  __shared__ float sm[256], ss[256];
  const int i = blockIdx.x, t = threadIdx.x;
  for (int j = t; j < NSRC; j += 256) lws[j] = Whst[j];
  __syncthreads();
  const int* __restrict__ Arow = A + (size_t)i * NSRC;
  const float rowc = Whtt[i];
  float m = NEGV, s = 0.f;
  for (int j = t; j < NSRC; j += 256) {
    float x = rowc + lws[j];
    float e = x > 0.f ? x : LRA * x;
    e = (Arow[j] > 0) ? e : NEGV;
    online_ms(m, s, e);
  }
  sm[t] = m; ss[t] = s;
  __syncthreads();
  for (int off = 128; off > 0; off >>= 1) {
    if (t < off) {
      float m1 = sm[t], s1 = ss[t];
      combine_ms(m1, s1, sm[t + off], ss[t + off]);
      sm[t] = m1; ss[t] = s1;
    }
    __syncthreads();
  }
  if (t == 0) {
    m_st[i] = sm[0];
    inv_st[i] = 1.f / ss[0];           // s >= 1 always (max element contributes 1)
  }
}

// ---------------------------------------------------------------------------
// Kernel 3: stats per row of att_ts (columns of A). Block owns 64 consecutive
// columns x all 4096 rows; 4 row-phases per column so loads stay coalesced.
// ---------------------------------------------------------------------------
__global__ void gat_stats_ts_kernel(const int* __restrict__ A, const float* __restrict__ Whst,
                                    const float* __restrict__ Whtt,
                                    float* __restrict__ m_ts, float* __restrict__ inv_ts) {
  __shared__ float lwt[NTGT];          // 16 KB
  __shared__ float sm[256], ss[256];
  const int t = threadIdx.x;
  const int j = blockIdx.x * 64 + (t & 63);
  const int ty = t >> 6;
  for (int i = t; i < NTGT; i += 256) lwt[i] = Whtt[i];
  __syncthreads();
  const float colc = Whst[j];
  float m = NEGV, s = 0.f;
  for (int i = ty; i < NTGT; i += 4) {
    float x = lwt[i] + colc;
    float e = x > 0.f ? x : LRA * x;
    e = (A[(size_t)i * NSRC + j] > 0) ? e : NEGV;
    online_ms(m, s, e);
  }
  sm[t] = m; ss[t] = s;
  __syncthreads();
  if (t < 128) {
    float m1 = sm[t], s1 = ss[t];
    combine_ms(m1, s1, sm[t + 128], ss[t + 128]);
    sm[t] = m1; ss[t] = s1;
  }
  __syncthreads();
  if (t < 64) {
    float m1 = sm[t], s1 = ss[t];
    combine_ms(m1, s1, sm[t + 64], ss[t + 64]);
    m_ts[j] = m1;
    inv_ts[j] = 1.f / s1;
  }
}

// ---------------------------------------------------------------------------
// Kernel 4: fused att_st @ Whs via v_wmma_f32_16x16x32_f16.
// Wave owns a 16x64 output tile; A-fragment (att weights) generated on the
// fly in the ISA's 16-bit A-matrix layout:
//   lanes 0-15 : M=lane,    slots 0-7 -> K=0..7,  slots 8-15 -> K=16..23
//   lanes 16-31: M=lane-16, slots 0-7 -> K=8..15, slots 8-15 -> K=24..31
// Weight generation is branchless: masked logits become NEGV, exp underflows
// to exactly 0, so no EXEC-mask branches appear in the hot loop.
// ---------------------------------------------------------------------------
__global__ __launch_bounds__(128)
void gat_attn_st_kernel(const int* __restrict__ A, const float* __restrict__ Whst,
                        const float* __restrict__ Whtt,
                        const float* __restrict__ m_st, const float* __restrict__ inv_st,
                        const _Float16* __restrict__ Bh, const float* __restrict__ bias,
                        float* __restrict__ out) {
  __shared__ float lws[NSRC];          // Whst staged in LDS (32 KB)
  const int t = threadIdx.x;
  for (int j = t; j < NSRC; j += 128) lws[j] = Whst[j];
  __syncthreads();
  const int wave = t >> 5, l = t & 31;
  const int hA = l >> 4, Ma = l & 15;
  const int i0 = blockIdx.x * 64 + wave * 16;
  const int i  = i0 + Ma;
  const int* __restrict__ Arow = A + (size_t)i * NSRC;
  const float rowc = Whtt[i];
  const float rmax = m_st[i];
  const v16h* __restrict__ Bf = (const v16h*)Bh;
  v8f c0 = {}, c1 = {}, c2 = {}, c3 = {};
#pragma unroll 1
  for (int k0 = 0; k0 < NSRC; k0 += 32) {
    const int kt = k0 >> 5;
    const int base0 = k0 + hA * 8;
    const int base1 = base0 + 16;
    v4i a0 = *(const v4i*)(Arow + base0);
    v4i a1v = *(const v4i*)(Arow + base0 + 4);
    v4i a2v = *(const v4i*)(Arow + base1);
    v4i a3v = *(const v4i*)(Arow + base1 + 4);
    v4f w0 = *(const v4f*)(lws + base0);
    v4f w1 = *(const v4f*)(lws + base0 + 4);
    v4f w2 = *(const v4f*)(lws + base1);
    v4f w3 = *(const v4f*)(lws + base1 + 4);
    v16h af;
#pragma unroll
    for (int s = 0; s < 4; ++s) {
      float x0 = rowc + w0[s]; x0 = x0 > 0.f ? x0 : LRA * x0;
      x0 = (a0[s]  > 0) ? x0 : NEGV;
      af[s]      = (_Float16)__expf(x0 - rmax);
      float x1 = rowc + w1[s]; x1 = x1 > 0.f ? x1 : LRA * x1;
      x1 = (a1v[s] > 0) ? x1 : NEGV;
      af[s + 4]  = (_Float16)__expf(x1 - rmax);
      float x2 = rowc + w2[s]; x2 = x2 > 0.f ? x2 : LRA * x2;
      x2 = (a2v[s] > 0) ? x2 : NEGV;
      af[s + 8]  = (_Float16)__expf(x2 - rmax);
      float x3 = rowc + w3[s]; x3 = x3 > 0.f ? x3 : LRA * x3;
      x3 = (a3v[s] > 0) ? x3 : NEGV;
      af[s + 12] = (_Float16)__expf(x3 - rmax);
    }
    v16h b0 = Bf[(size_t)(kt * 4 + 0) * 32 + l];
    v16h b1 = Bf[(size_t)(kt * 4 + 1) * 32 + l];
    v16h b2 = Bf[(size_t)(kt * 4 + 2) * 32 + l];
    v16h b3 = Bf[(size_t)(kt * 4 + 3) * 32 + l];
    c0 = __builtin_amdgcn_wmma_f32_16x16x32_f16(false, af, false, b0, (short)0, c0, false, false);
    c1 = __builtin_amdgcn_wmma_f32_16x16x32_f16(false, af, false, b1, (short)0, c1, false, false);
    c2 = __builtin_amdgcn_wmma_f32_16x16x32_f16(false, af, false, b2, (short)0, c2, false, false);
    c3 = __builtin_amdgcn_wmma_f32_16x16x32_f16(false, af, false, b3, (short)0, c3, false, false);
  }
  // C/D layout: lane half h, VGPR v -> row M = v + 8*h, col N = lane&15.
  const int col0 = l & 15;
#pragma unroll
  for (int v = 0; v < 8; ++v) {
    int ri = i0 + v + 8 * hA;
    float iv = inv_st[ri];
    float* orow = out + (size_t)ri * DOUT;
    orow[col0]      = elu_bf(c0[v] * iv + bias[col0]);
    orow[col0 + 16] = elu_bf(c1[v] * iv + bias[col0 + 16]);
    orow[col0 + 32] = elu_bf(c2[v] * iv + bias[col0 + 32]);
    orow[col0 + 48] = elu_bf(c3[v] * iv + bias[col0 + 48]);
  }
}

// ---------------------------------------------------------------------------
// Kernel 5: fused att_ts @ Wht (transposed mask access: att_ts[j,i] uses
// A[i,j]); same WMMA scheme with K running over targets. A fits in the
// 192 MB L2 so the column-strided reads are L2-served.
// ---------------------------------------------------------------------------
__global__ __launch_bounds__(128)
void gat_attn_ts_kernel(const int* __restrict__ A, const float* __restrict__ Whst,
                        const float* __restrict__ Whtt,
                        const float* __restrict__ m_ts, const float* __restrict__ inv_ts,
                        const _Float16* __restrict__ Bh, const float* __restrict__ bias,
                        float* __restrict__ out) {
  __shared__ float lwt[NTGT];          // Whtt staged in LDS (16 KB)
  const int t = threadIdx.x;
  for (int i = t; i < NTGT; i += 128) lwt[i] = Whtt[i];
  __syncthreads();
  const int wave = t >> 5, l = t & 31;
  const int hA = l >> 4, Ma = l & 15;
  const int j0 = blockIdx.x * 64 + wave * 16;
  const int j  = j0 + Ma;
  const float colc = Whst[j];
  const float rmax = m_ts[j];
  const v16h* __restrict__ Bf = (const v16h*)Bh;
  v8f c0 = {}, c1 = {}, c2 = {}, c3 = {};
#pragma unroll 1
  for (int k0 = 0; k0 < NTGT; k0 += 32) {
    const int kt = k0 >> 5;
    const int kb0 = k0 + hA * 8;
    const int kb1 = kb0 + 16;
    v16h af;
#pragma unroll
    for (int s = 0; s < 8; ++s) {
      int ia = kb0 + s;
      int av = A[(size_t)ia * NSRC + j];
      float x = lwt[ia] + colc; x = x > 0.f ? x : LRA * x;
      x = (av > 0) ? x : NEGV;
      af[s] = (_Float16)__expf(x - rmax);
      int ib = kb1 + s;
      int av2 = A[(size_t)ib * NSRC + j];
      float x2 = lwt[ib] + colc; x2 = x2 > 0.f ? x2 : LRA * x2;
      x2 = (av2 > 0) ? x2 : NEGV;
      af[s + 8] = (_Float16)__expf(x2 - rmax);
    }
    v16h b0 = Bf[(size_t)(kt * 4 + 0) * 32 + l];
    v16h b1 = Bf[(size_t)(kt * 4 + 1) * 32 + l];
    v16h b2 = Bf[(size_t)(kt * 4 + 2) * 32 + l];
    v16h b3 = Bf[(size_t)(kt * 4 + 3) * 32 + l];
    c0 = __builtin_amdgcn_wmma_f32_16x16x32_f16(false, af, false, b0, (short)0, c0, false, false);
    c1 = __builtin_amdgcn_wmma_f32_16x16x32_f16(false, af, false, b1, (short)0, c1, false, false);
    c2 = __builtin_amdgcn_wmma_f32_16x16x32_f16(false, af, false, b2, (short)0, c2, false, false);
    c3 = __builtin_amdgcn_wmma_f32_16x16x32_f16(false, af, false, b3, (short)0, c3, false, false);
  }
  const int col0 = l & 15;
#pragma unroll
  for (int v = 0; v < 8; ++v) {
    int rj = j0 + v + 8 * hA;
    float iv = inv_ts[rj];
    float* orow = out + (size_t)rj * DOUT;
    orow[col0]      = elu_bf(c0[v] * iv + bias[col0]);
    orow[col0 + 16] = elu_bf(c1[v] * iv + bias[col0 + 16]);
    orow[col0 + 32] = elu_bf(c2[v] * iv + bias[col0 + 32]);
    orow[col0 + 48] = elu_bf(c3[v] * iv + bias[col0 + 48]);
  }
}

// ---------------------------------------------------------------------------
extern "C" void kernel_launch(void* const* d_in, const int* in_sizes, int n_in,
                              void* d_out, int out_size, void* d_ws, size_t ws_size,
                              hipStream_t stream) {
  (void)in_sizes; (void)n_in; (void)out_size; (void)ws_size;
  const float* hs     = (const float*)d_in[0];
  const float* ht     = (const float*)d_in[1];
  const int*   A      = (const int*)d_in[2];
  const float* Ws     = (const float*)d_in[3];
  const float* Wt     = (const float*)d_in[4];
  const float* a1     = (const float*)d_in[5];
  const float* bias_s = (const float*)d_in[6];
  const float* bias_t = (const float*)d_in[7];

  char* w = (char*)d_ws;
  float* Whs    = (float*)w; w += (size_t)NSRC * DOUT * 4;
  float* Wht    = (float*)w; w += (size_t)NTGT * DOUT * 4;
  float* Whst   = (float*)w; w += (size_t)NSRC * 4;
  float* Whtt   = (float*)w; w += (size_t)NTGT * 4;
  float* m_st   = (float*)w; w += (size_t)NTGT * 4;
  float* inv_st = (float*)w; w += (size_t)NTGT * 4;
  float* m_ts   = (float*)w; w += (size_t)NSRC * 4;
  float* inv_ts = (float*)w; w += (size_t)NSRC * 4;
  _Float16* Whs_h = (_Float16*)w; w += (size_t)NSRC * DOUT * 2;
  _Float16* Wht_h = (_Float16*)w; w += (size_t)NTGT * DOUT * 2;

  float* out_st = (float*)d_out;                      // (4096, 64) elu(h_prime_st)
  float* out_ts = out_st + (size_t)NTGT * DOUT;       // (8192, 64) elu(h_prime_ts)

  gat_proj_kernel<<<NSRC + NTGT, 64, 0, stream>>>(hs, ht, Ws, Wt, a1, Whs, Wht, Whst, Whtt);
  gat_pack_kernel<<<(NSRC * DOUT) / 256, 256, 0, stream>>>(Whs, Whs_h, NSRC);
  gat_pack_kernel<<<(NTGT * DOUT) / 256, 256, 0, stream>>>(Wht, Wht_h, NTGT);
  gat_stats_st_kernel<<<NTGT, 256, 0, stream>>>(A, Whst, Whtt, m_st, inv_st);
  gat_stats_ts_kernel<<<NSRC / 64, 256, 0, stream>>>(A, Whst, Whtt, m_ts, inv_ts);
  gat_attn_st_kernel<<<NTGT / 64, 128, 0, stream>>>(A, Whst, Whtt, m_st, inv_st, Whs_h, bias_s, out_st);
  gat_attn_ts_kernel<<<NSRC / 64, 128, 0, stream>>>(A, Whst, Whtt, m_ts, inv_ts, Wht_h, bias_t, out_ts);
}
